// PointnetLFPModuleMSG_5669356832938
// MI455X (gfx1250) — compile-verified, matrix operand order checked by reading, not run on hardware
//
#include <hip/hip_runtime.h>
#include <hip/hip_bf16.h>

#define BC   4
#define N1C  4096
#define N2C  4096
#define C1C  64

typedef _Float16 h16 __attribute__((ext_vector_type(16)));
typedef float    f8  __attribute__((ext_vector_type(8)));
typedef float    f4v __attribute__((ext_vector_type(4)));

// Load one 16-half WMMA fragment as two 16-byte chunks.
// Per CDNA5 ISA (CDNA5 ISA): lane<16 holds K={o..o+7, o+16..o+23}, lane>=16 holds
// K={o+8..o+15, o+24..o+31}; caller passes the per-lane offsets o0, o1 = o0+16.
__device__ inline h16 frag_ld(const _Float16* p, int o0, int o1) {
    h16 r;
    *reinterpret_cast<f4v*>(&r)       = *reinterpret_cast<const f4v*>(p + o0);
    *(reinterpret_cast<f4v*>(&r) + 1) = *reinterpret_cast<const f4v*>(p + o1);
    return r;
}

// ---------------------------------------------------------------------------
// Prep: transpose (B,C,N) f32 -> (B,N,C) f16 rows (contiguous per point)
// ---------------------------------------------------------------------------
__global__ void transpose_to_f16(const float* __restrict__ src,
                                 _Float16* __restrict__ dst, int C, int N) {
    int t = blockIdx.x * blockDim.x + threadIdx.x;   // flat dst index, c fastest
    int c = t % C;
    int rem = t / C;
    int n = rem % N;
    int b = rem / N;
    dst[t] = (_Float16)src[((size_t)b * C + c) * N + n];
}

// layer0 weights (64,67) -> f16 (64 x 96): cols 0..63 = feat weights (w[:,3:67]),
// cols 64..66 = xyz weights (w[:,0:3]), cols 67..95 = 0  (matches g-matrix layout)
__global__ void prep_w0(const float* __restrict__ w, _Float16* __restrict__ dst) {
    int t = blockIdx.x * blockDim.x + threadIdx.x;   // 64*96
    int n = t / 96, k = t % 96;
    float v = 0.f;
    if (k < 64)      v = w[n * 67 + 3 + k];
    else if (k < 67) v = w[n * 67 + (k - 64)];
    dst[t] = (_Float16)v;
}

__global__ void cvt_f16(const float* __restrict__ s, _Float16* __restrict__ d, int n) {
    int t = blockIdx.x * blockDim.x + threadIdx.x;
    if (t < n) d[t] = (_Float16)s[t];
}

// ---------------------------------------------------------------------------
// Ball query: one wave32 per query; ordered first-k extraction via ballot.
// Emits idx0 (16 per query, r=0.1) and idx1 (32 per query, r=0.2).
// ---------------------------------------------------------------------------
__global__ __launch_bounds__(256) void ball_query_kernel(
    const float* __restrict__ xyz2, const float* __restrict__ xyz1,
    int* __restrict__ idx0, int* __restrict__ idx1) {
    const int lane = threadIdx.x & 31;
    const int gq = blockIdx.x * 8 + (threadIdx.x >> 5);   // 0..B*N2-1
    const int b = gq / N2C;
    const float* p2 = xyz2 + (size_t)gq * 3;
    const float px = p2[0], py = p2[1], pz = p2[2];

    int* o0 = idx0 + (size_t)gq * 16;
    int* o1 = idx1 + (size_t)gq * 32;
    int cnt0 = 0, cnt1 = 0, first0 = -1, first1 = -1;
    const unsigned pre = (1u << lane) - 1u;

    for (int base = 0; base < N1C; base += 32) {
        if (cnt0 >= 16 && cnt1 >= 32) break;
        const float* p1 = xyz1 + ((size_t)b * N1C + base + lane) * 3;
        float dx = p1[0] - px, dy = p1[1] - py, dz = p1[2] - pz;
        float d2 = dx * dx + dy * dy + dz * dz;
        unsigned msk1 = __builtin_amdgcn_ballot_w32(d2 < 0.04f);   // r=0.2
        unsigned msk0 = __builtin_amdgcn_ballot_w32(d2 < 0.01f);   // r=0.1
        if (cnt0 < 16) {
            int pos = cnt0 + __popc(msk0 & pre);
            if (((msk0 >> lane) & 1u) && pos < 16) o0[pos] = base + lane;
            if (first0 < 0 && msk0) first0 = base + __ffs(msk0) - 1;
            cnt0 = min(cnt0 + __popc(msk0), 16);
        }
        if (cnt1 < 32) {
            int pos = cnt1 + __popc(msk1 & pre);
            if (((msk1 >> lane) & 1u) && pos < 32) o1[pos] = base + lane;
            if (first1 < 0 && msk1) first1 = base + __ffs(msk1) - 1;
            cnt1 = min(cnt1 + __popc(msk1), 32);
        }
    }
    // pad with first valid index (0 if none) — matches reference semantics
    if (lane < 16 && lane >= cnt0) o0[lane] = (cnt0 > 0) ? first0 : 0;
    if (lane >= cnt1)              o1[lane] = (cnt1 > 0) ? first1 : 0;
}

// ---------------------------------------------------------------------------
// Grouped MLP + maxpool for one branch. 128 sample-rows / workgroup, 8 waves.
// g-matrix cols: 0..63 gathered feats, 64..66 relative xyz, 67..95 zero pad.
// ---------------------------------------------------------------------------
template <int S>
__global__ __launch_bounds__(256, 1) void group_mlp_kernel(
    const float* __restrict__ xyz2, const float* __restrict__ xyz1,
    const _Float16* __restrict__ feats1h, const int* __restrict__ idxA,
    const _Float16* __restrict__ w0h, const float* __restrict__ b0,
    const _Float16* __restrict__ w1h, const float* __restrict__ b1,
    _Float16* __restrict__ poolh) {
    constexpr int QB = 128 / S;     // queries per workgroup
    constexpr int G  = S / 16;      // M-tiles per query

    __shared__ __align__(16) _Float16 gmat[128 * 96];
    __shared__ __align__(16) _Float16 hmat[128 * 64];
    __shared__ __align__(16) _Float16 w0s[64 * 96];
    __shared__ __align__(16) _Float16 w1s[64 * 64];
    __shared__ float pools[8][64];

    const int tid = threadIdx.x;
    const int qPerB = N2C / QB;
    const int b  = blockIdx.x / qPerB;
    const int q0 = (blockIdx.x % qPerB) * QB;

    // weights -> LDS (2 halfs per load)
    {
        const unsigned* s0 = (const unsigned*)w0h;
        unsigned* d0 = (unsigned*)w0s;
        for (int i = tid; i < 64 * 96 / 2; i += 256) d0[i] = s0[i];
        const unsigned* s1 = (const unsigned*)w1h;
        unsigned* d1 = (unsigned*)w1s;
        for (int i = tid; i < 64 * 64 / 2; i += 256) d1[i] = s1[i];
    }

    // gather g-matrix: 2 threads per row
    {
        int r = tid >> 1, hf = tid & 1;
        int q = q0 + r / S, s = r % S;
        int gq = b * N2C + q;
        int idx = idxA[(size_t)gq * S + s];
        const f4v* src = reinterpret_cast<const f4v*>(
                             feats1h + ((size_t)b * N1C + idx) * 64) + hf * 4;
        f4v* dst = reinterpret_cast<f4v*>(gmat + r * 96) + hf * 4;
        dst[0] = src[0]; dst[1] = src[1]; dst[2] = src[2]; dst[3] = src[3];
        if (hf == 0) {
            const float* p1 = xyz1 + ((size_t)b * N1C + idx) * 3;
            const float* p2 = xyz2 + (size_t)gq * 3;
            gmat[r * 96 + 64] = (_Float16)(p1[0] - p2[0]);
            gmat[r * 96 + 65] = (_Float16)(p1[1] - p2[1]);
            gmat[r * 96 + 66] = (_Float16)(p1[2] - p2[2]);
        } else {
            for (int c = 67; c < 96; ++c) gmat[r * 96 + c] = (_Float16)0.f;
        }
    }
    __syncthreads();

    const int wid  = tid >> 5, lane = tid & 31;
    const int m0   = wid * 16;
    const int lrow = lane & 15;
    const int kh   = lane >> 4;
    const int mrow = m0 + lrow;

    // ---- layer 0: (128x96) x (96x64), K tiles of 32 ----
    f8 acc0[4] = {};
    for (int kb = 0; kb < 3; ++kb) {
        int o0 = kb * 32 + kh * 8, o1 = o0 + 16;
        h16 a = frag_ld(gmat + mrow * 96, o0, o1);
#pragma unroll
        for (int nt = 0; nt < 4; ++nt) {
            h16 bb = frag_ld(w0s + (nt * 16 + lrow) * 96, o0, o1);
            acc0[nt] = __builtin_amdgcn_wmma_f32_16x16x32_f16(
                false, a, false, bb, (short)0, acc0[nt], false, false);
        }
    }
#pragma unroll
    for (int nt = 0; nt < 4; ++nt) {
        int n = nt * 16 + lrow;
        float bv = b0[n];
#pragma unroll
        for (int r = 0; r < 8; ++r) {
            float v = fmaxf(acc0[nt][r] + bv, 0.f);
            hmat[(m0 + r + 8 * kh) * 64 + n] = (_Float16)v;   // D layout: M = r + 8*kh
        }
    }
    __syncthreads();

    // ---- layer 1: (128x64) x (64x64) ----
    f8 acc1[4] = {};
    for (int kb = 0; kb < 2; ++kb) {
        int o0 = kb * 32 + kh * 8, o1 = o0 + 16;
        h16 a = frag_ld(hmat + mrow * 64, o0, o1);
#pragma unroll
        for (int nt = 0; nt < 4; ++nt) {
            h16 bb = frag_ld(w1s + (nt * 16 + lrow) * 64, o0, o1);
            acc1[nt] = __builtin_amdgcn_wmma_f32_16x16x32_f16(
                false, a, false, bb, (short)0, acc1[nt], false, false);
        }
    }
    // bias + relu + max over 16 rows of this M-tile
#pragma unroll
    for (int nt = 0; nt < 4; ++nt) {
        float bv = b1[nt * 16 + lrow];
        float vm = 0.f;                       // relu floor, so 0 is a safe identity
#pragma unroll
        for (int r = 0; r < 8; ++r) vm = fmaxf(vm, acc1[nt][r] + bv);
        vm = fmaxf(vm, __shfl_xor(vm, 16, 32));   // combine M halves across lanes
        if (lane < 16) pools[wid][nt * 16 + lane] = vm;
    }
    __syncthreads();

    // final max across M-tiles of each query; write pooled f16
    for (int t = tid; t < QB * 64; t += 256) {
        int lq = t >> 6, ch = t & 63;
        float vm = pools[lq * G][ch];
        if (G == 2) vm = fmaxf(vm, pools[lq * G + 1][ch]);
        poolh[((size_t)(b * N2C + q0 + lq)) * 64 + ch] = (_Float16)vm;
    }
}

// ---------------------------------------------------------------------------
// Post MLP: y = relu([pooled | f2t] @ post_w^T + b), written transposed to out
// 128 queries per workgroup; weights streamed from global (L2-resident 32KB).
// ---------------------------------------------------------------------------
__global__ __launch_bounds__(256, 1) void post_kernel(
    const _Float16* __restrict__ poolh, const _Float16* __restrict__ f2h,
    const _Float16* __restrict__ pwh, const float* __restrict__ pb,
    float* __restrict__ out, int chOff) {
    __shared__ __align__(16) _Float16 xs[128 * 128];

    const int tid = threadIdx.x;
    const int qPerB = N2C / 128;
    const int b  = blockIdx.x / qPerB;
    const int q0 = (blockIdx.x % qPerB) * 128;

    // stage x rows: [pooled(64) | f2t(64)] as f16
    for (int i = tid; i < 128 * 16; i += 256) {
        int r = i >> 4, seg = i & 15;          // 16 x 16B chunks per 128-half row
        size_t gq = (size_t)b * N2C + q0 + r;
        const f4v* src = (seg < 8)
            ? reinterpret_cast<const f4v*>(poolh + gq * 64) + seg
            : reinterpret_cast<const f4v*>(f2h + gq * 64) + (seg - 8);
        reinterpret_cast<f4v*>(xs)[i] = *src;
    }
    __syncthreads();

    const int wid  = tid >> 5, lane = tid & 31;
    const int lrow = lane & 15, kh = lane >> 4;
    const int mrow = wid * 16 + lrow;

    f8 acc[8] = {};
    for (int kb = 0; kb < 4; ++kb) {
        int o0 = kb * 32 + kh * 8, o1 = o0 + 16;
        h16 a = frag_ld(xs + mrow * 128, o0, o1);
#pragma unroll
        for (int nt = 0; nt < 8; ++nt) {
            h16 bb = frag_ld(pwh + (nt * 16 + lrow) * 128, o0, o1);
            acc[nt] = __builtin_amdgcn_wmma_f32_16x16x32_f16(
                false, a, false, bb, (short)0, acc[nt], false, false);
        }
    }
#pragma unroll
    for (int nt = 0; nt < 8; ++nt) {
        int n = nt * 16 + lrow;
        float bv = pb[n];
#pragma unroll
        for (int r = 0; r < 8; ++r) {
            float v = fmaxf(acc[nt][r] + bv, 0.f);
            int q = q0 + wid * 16 + r + 8 * kh;
            out[((size_t)b * 256 + chOff + n) * N2C + q] = v;
        }
    }
}

// ---------------------------------------------------------------------------
extern "C" void kernel_launch(void* const* d_in, const int* in_sizes, int n_in,
                              void* d_out, int out_size, void* d_ws, size_t ws_size,
                              hipStream_t stream) {
    (void)in_sizes; (void)n_in; (void)out_size; (void)ws_size;
    const float* xyz2      = (const float*)d_in[0];
    const float* xyz1      = (const float*)d_in[1];
    const float* features2 = (const float*)d_in[2];
    const float* features1 = (const float*)d_in[3];
    const float* m0w0 = (const float*)d_in[4];
    const float* m0b0 = (const float*)d_in[5];
    const float* m0w1 = (const float*)d_in[6];
    const float* m0b1 = (const float*)d_in[7];
    const float* m1w0 = (const float*)d_in[8];
    const float* m1b0 = (const float*)d_in[9];
    const float* m1w1 = (const float*)d_in[10];
    const float* m1b1 = (const float*)d_in[11];
    const float* pw   = (const float*)d_in[12];
    const float* pb   = (const float*)d_in[13];
    float* out = (float*)d_out;

    char* ws = (char*)d_ws;
    size_t off = 0;
    auto alloc = [&](size_t bytes) -> void* {
        void* p = ws + off;
        off = (off + bytes + 255) & ~(size_t)255;
        return p;
    };
    _Float16* feats1h = (_Float16*)alloc((size_t)BC * N1C * 64 * 2);
    _Float16* f2h     = (_Float16*)alloc((size_t)BC * N2C * 64 * 2);
    int*      idx0    = (int*)alloc((size_t)BC * N2C * 16 * 4);
    int*      idx1    = (int*)alloc((size_t)BC * N2C * 32 * 4);
    _Float16* w0h0    = (_Float16*)alloc(64 * 96 * 2);
    _Float16* w0h1    = (_Float16*)alloc(64 * 96 * 2);
    _Float16* w1h0    = (_Float16*)alloc(64 * 64 * 2);
    _Float16* w1h1    = (_Float16*)alloc(64 * 64 * 2);
    _Float16* pwh     = (_Float16*)alloc(128 * 128 * 2);
    _Float16* pool0   = (_Float16*)alloc((size_t)BC * N2C * 64 * 2);
    _Float16* pool1   = (_Float16*)alloc((size_t)BC * N2C * 64 * 2);

    // prep
    transpose_to_f16<<<BC * N1C * 64 / 256, 256, 0, stream>>>(features1, feats1h, 64, N1C);
    transpose_to_f16<<<BC * N2C * 64 / 256, 256, 0, stream>>>(features2, f2h, 64, N2C);
    prep_w0<<<24, 256, 0, stream>>>(m0w0, w0h0);
    prep_w0<<<24, 256, 0, stream>>>(m1w0, w0h1);
    cvt_f16<<<16, 256, 0, stream>>>(m0w1, w1h0, 64 * 64);
    cvt_f16<<<16, 256, 0, stream>>>(m1w1, w1h1, 64 * 64);
    cvt_f16<<<64, 256, 0, stream>>>(pw, pwh, 128 * 128);

    // ball query (both radii in one pass)
    ball_query_kernel<<<BC * N2C / 8, 256, 0, stream>>>(xyz2, xyz1, idx0, idx1);

    // branch MLPs + maxpool
    group_mlp_kernel<16><<<BC * N2C / 8, 256, 0, stream>>>(
        xyz2, xyz1, feats1h, idx0, w0h0, m0b0, w1h0, m0b1, pool0);
    group_mlp_kernel<32><<<BC * N2C / 4, 256, 0, stream>>>(
        xyz2, xyz1, feats1h, idx1, w0h1, m1b0, w1h1, m1b1, pool1);

    // post MLP, transposed writes into (B, 256, N2) output
    post_kernel<<<BC * N2C / 128, 256, 0, stream>>>(pool0, f2h, pwh, pb, out, 0);
    post_kernel<<<BC * N2C / 128, 256, 0, stream>>>(pool1, f2h, pwh, pb, out, 128);
}
